// GARCHGRUModel_83605833383958
// MI455X (gfx1250) — matrix operand order repaired and checked by reading, not compiled, not run on hardware
//
#include <hip/hip_runtime.h>

// ---------------------------------------------------------------------------
// GARCH-GRU forward for MI455X (gfx1250): bf16 WMMA recurrence, batch-split.
// B=512, T=256, H=256, HOR=22, IN=1.
// Round 4: packed-B loads as true global_load_b128 (addrspace(1) cast) with
// per-iteration laundered *offset* so LICM still cannot hoist them.
// ---------------------------------------------------------------------------

typedef __bf16 bf16;
typedef __attribute__((ext_vector_type(16))) __bf16 v16bf;
typedef __attribute__((ext_vector_type(8)))  float  v8f;
typedef __attribute__((ext_vector_type(4)))  unsigned int u32x4;

typedef __attribute__((address_space(1))) const u32x4 gc_u32x4;

#define BSZ 512
#define TT  256
#define HH  256
#define HOR 22
#define KCH 8   // K chunks of 32 (256/32)

// ---- fragment helpers ------------------------------------------------------

union FragU {
  struct { u32x4 lo, hi; } u;
  v16bf v;
};

static __device__ __forceinline__ v16bf make_frag(u32x4 lo, u32x4 hi) {
  FragU f; f.u.lo = lo; f.u.hi = hi; return f.v;
}

// Opaque zero offset: defeats LICM (loads whose address depends on it cannot
// be hoisted out of the loop that re-evaluates it) without obscuring the
// base pointer's address space.
static __device__ __forceinline__ int launder_off() {
  int off = 0;
  asm volatile("" : "+v"(off));
  return off;
}

// Per-lane base pointer into an LDS row-major [16][256] bf16 tile for the
// WMMA A fragment: lanes 0-15 rows M=0..15 hold K {0..7,16..23}; lanes 16-31
// hold K {8..15,24..31}.
static __device__ __forceinline__ const bf16* a_base(const bf16* sm, int lane) {
  return sm + (lane & 15) * HH + ((lane >> 4) << 3);
}
static __device__ __forceinline__ v16bf load_a(const bf16* base, int kc) {
  const bf16* p = base + kc * 32;
  u32x4 lo = *(const u32x4*)(p);
  u32x4 hi = *(const u32x4*)(p + 16);
  return make_frag(lo, hi);
}

// Per-lane base pointer into a packed B buffer for N-tile nt.
// Fragment (nt,kc) element address: ((nt*KCH+kc)*32 + lane)*16.
static __device__ __forceinline__ const bf16* b_base(const bf16* pk, int nt, int lane) {
  return pk + (nt * KCH * 32 + lane) * 16;
}
// Global (addrspace 1) fragment load -> global_load_b128, LOADcnt only.
static __device__ __forceinline__ v16bf load_b(const bf16* base, int kc) {
  const gc_u32x4* g = (const gc_u32x4*)(base + kc * 512);
  u32x4 lo = g[0];
  u32x4 hi = g[1];
  return make_frag(lo, hi);
}

static __device__ __forceinline__ v8f wmma_bf16(v16bf a, v16bf b, v8f c) {
  return __builtin_amdgcn_wmma_f32_16x16x32_bf16(false, a, false, b, (short)0, c,
                                                 false, false);
}

// ---- scalar math -----------------------------------------------------------

static __device__ __forceinline__ float sigmf(float v)     { return 1.0f / (1.0f + expf(-v)); }
static __device__ __forceinline__ float softplusf(float v) { return (v > 20.0f) ? v : log1pf(expf(v)); }

// ---------------------------------------------------------------------------
// Repack a row-major f32 [K][Nsrc] matrix into bf16 WMMA-B fragments,
// zero-padding columns up to Npad.
// ---------------------------------------------------------------------------
__global__ void garchgru_repack(const float* __restrict__ src, bf16* __restrict__ dst,
                                int K, int Nsrc, int Npad) {
  int idx = blockIdx.x * blockDim.x + threadIdx.x;
  int total = K * Npad;
  if (idx >= total) return;
  int j    = idx & 15;
  int lane = (idx >> 4) & 31;
  int f    = idx >> 9;                 // fragment index = nt*kchunks + kc
  int kchunks = K >> 5;
  int nt = f / kchunks;
  int kc = f - nt * kchunks;
  int col = nt * 16 + (lane & 15);
  int row = kc * 32 + ((lane >> 4) << 3) + (j & 7) + ((j >> 3) << 4);
  float v = (col < Nsrc) ? src[row * Nsrc + col] : 0.0f;
  dst[idx] = (bf16)v;
}

// ---------------------------------------------------------------------------
// Main persistent recurrence kernel: one workgroup per 16 batch rows,
// 16 waves x 32 lanes; wave w owns output columns [w*16, w*16+16).
// ---------------------------------------------------------------------------
__global__ __launch_bounds__(512)
void garchgru_main(const float* __restrict__ x,
                   const float* __restrict__ Wzw, const float* __restrict__ Wzb,
                   const float* __restrict__ Uzb,
                   const float* __restrict__ Wrw, const float* __restrict__ Wrb,
                   const float* __restrict__ Urb,
                   const float* __restrict__ Whw, const float* __restrict__ Whb,
                   const float* __restrict__ Uhb,
                   const float* __restrict__ Wgw, const float* __restrict__ Wgb,
                   const float* __restrict__ omega_raw, const float* __restrict__ alpha_raw,
                   const float* __restrict__ beta_raw, const float* __restrict__ gamma_p,
                   const float* __restrict__ fc1b, const float* __restrict__ fc2b,
                   const bf16* __restrict__ uzpk, const bf16* __restrict__ urpk,
                   const bf16* __restrict__ uhpk, const bf16* __restrict__ fc1pk,
                   const bf16* __restrict__ fc2pk,
                   float* __restrict__ out) {
  __shared__ bf16  sh_h[16 * HH];   // h_t, bf16, row-major (A staging)
  __shared__ bf16  sh_rh[16 * HH];  // r*h staging / hid staging
  __shared__ float sh_xt[16], sh_g[16], sh_eps[16], sh_sig[16];

  const int tid  = threadIdx.x;
  const int wave = tid >> 5;                    // 0..15, one N-tile each
  const int lane = tid & 31;
  const int nlo  = lane & 15;
  const int moff = (lane >> 4) << 3;            // 0 or 8 (D-matrix row offset)
  const int b0   = blockIdx.x * 16;
  const int nt   = wave;
  const int n    = nt * 16 + nlo;

  // Per-lane base pointers (all GEMM loads become base + immediate offset).
  const bf16* ah_base  = a_base(sh_h, lane);
  const bf16* arh_base = a_base(sh_rh, lane);
  const bf16* bz = b_base(uzpk, nt, lane);
  const bf16* br = b_base(urpk, nt, lane);
  const bf16* bh = b_base(uhpk, nt, lane);

  // GARCH scalars (uniform -> SGPRs)
  const float om  = softplusf(omega_raw[0]) + 1e-6f;
  const float aa  = sigmf(alpha_raw[0]);
  const float bb  = sigmf(beta_raw[0]) * (1.0f - aa * 0.99f);
  const float gam = gamma_p[0];

  // Per-column affine constants (fold IN=1 input projections + both biases).
  const float wz = Wzw[n], cz = Wzb[n] + Uzb[n];
  const float wr = Wrw[n], cr = Wrb[n] + Urb[n];
  const float wh = Whw[n], ch = Whb[n] + Uhb[n];
  const float wg = gam * Wgw[n], cg = gam * Wgb[n];

  // Own h values live in registers (owner-computes per (m,n) slot).
  float hr[8];
  #pragma unroll
  for (int j = 0; j < 8; ++j) hr[j] = 0.0f;

  for (int i = tid; i < 16 * HH; i += 512) sh_h[i] = (bf16)0.0f;
  if (tid < 16) { sh_eps[tid] = 1e-6f; sh_sig[tid] = 1e-6f; }
  __syncthreads();

  for (int t = 0; t < TT; ++t) {
    // --- GARCH scalar recursion + x_t staging (wave 0) ---
    if (wave == 0 && lane < 16) {
      float e = x[(b0 + lane) * TT + t];
      float g = om + aa * sh_eps[lane] + bb * sh_sig[lane];
      sh_xt[lane]  = e;
      sh_g[lane]   = g;
      sh_eps[lane] = e * e;
      sh_sig[lane] = g;
    }

    // Loop-variant opaque offset: pins all packed-B loads inside the t-loop.
    const int loff = launder_off();

    // --- z and r GEMMs: h @ Uz, h @ Ur (B double-buffered) ---
    v8f az = {}, ar = {};
    {
      const bf16* pz = bz + loff;
      const bf16* pr = br + loff;
      v16bf fz = load_b(pz, 0), fr = load_b(pr, 0);
      #pragma unroll
      for (int kc = 0; kc < KCH; ++kc) {
        const int kn = (kc + 1 < KCH) ? kc + 1 : kc;
        v16bf fzn = load_b(pz, kn);
        v16bf frn = load_b(pr, kn);
        v16bf a = load_a(ah_base, kc);
        az = wmma_bf16(a, fz, az);
        ar = wmma_bf16(a, fr, ar);
        fz = fzn; fr = frn;
        __builtin_amdgcn_sched_barrier(0);   // cap live ranges per iteration
      }
    }
    __syncthreads();  // x_t/garch visible; all sh_h reads for this step done

    float xtr[8], gar[8];
    #pragma unroll
    for (int j = 0; j < 8; ++j) { xtr[j] = sh_xt[j + moff]; gar[j] = sh_g[j + moff]; }

    // --- elementwise z, r; stage r*h in bf16 for the Uh GEMM ---
    #pragma unroll
    for (int j = 0; j < 8; ++j) {
      int m = j + moff;
      float zv = sigmf(az[j] + xtr[j] * wz + cz);
      float rv = sigmf(ar[j] + xtr[j] * wr + cr);
      az[j] = zv;                       // keep z in the accumulator registers
      sh_rh[m * HH + n] = (bf16)(rv * hr[j]);
    }
    __syncthreads();  // r*h staged

    // --- candidate GEMM: (r*h) @ Uh ---
    v8f ahh = {};
    {
      const bf16* ph = bh + loff;
      v16bf f = load_b(ph, 0);
      #pragma unroll
      for (int kc = 0; kc < KCH; ++kc) {
        const int kn = (kc + 1 < KCH) ? kc + 1 : kc;
        v16bf fn = load_b(ph, kn);
        v16bf a = load_a(arh_base, kc);
        ahh = wmma_bf16(a, f, ahh);
        f = fn;
        __builtin_amdgcn_sched_barrier(0);
      }
    }

    // --- gate combine + GARCH injection + tanh; publish h_{t+1} ---
    #pragma unroll
    for (int j = 0; j < 8; ++j) {
      int m = j + moff;
      float ht = tanhf(ahh[j] + xtr[j] * wh + ch);
      float zv = az[j];
      float hh = (1.0f - zv) * ht + zv * hr[j];
      float hn = tanhf(hh + gar[j] * wg + cg);
      hr[j] = hn;
      sh_h[m * HH + n] = (bf16)hn;
    }
    __syncthreads();  // h_{t+1} published before next step's GEMMs
  }

  // --- head: hid = relu(h @ fc1 + b1) ---
  const float f1bv = fc1b[n];
  const bf16* bf1 = b_base(fc1pk, nt, lane);
  v8f af = {};
  {
    v16bf f = load_b(bf1, 0);
    #pragma unroll
    for (int kc = 0; kc < KCH; ++kc) {
      const int kn = (kc + 1 < KCH) ? kc + 1 : kc;
      v16bf fn = load_b(bf1, kn);
      v16bf a = load_a(ah_base, kc);
      af = wmma_bf16(a, f, af);
      f = fn;
      __builtin_amdgcn_sched_barrier(0);
    }
  }
  #pragma unroll
  for (int j = 0; j < 8; ++j) {
    int m = j + moff;
    sh_rh[m * HH + n] = (bf16)fmaxf(af[j] + f1bv, 0.0f);
  }
  __syncthreads();

  // --- nn_scale = softplus(hid @ fc2 + b2); vol = clip(vol_base*(1+scale)) ---
  if (wave < 2) {   // fc2 is 256x22 padded to 32 cols -> N-tiles 0 and 1
    const float f2bv = (n < HOR) ? fc2b[n] : 0.0f;
    const bf16* bf2 = b_base(fc2pk, nt, lane);
    v8f s = {};
    {
      v16bf f = load_b(bf2, 0);
      #pragma unroll
      for (int kc = 0; kc < KCH; ++kc) {
        const int kn = (kc + 1 < KCH) ? kc + 1 : kc;
        v16bf fn = load_b(bf2, kn);
        v16bf a = load_a(arh_base, kc);
        s = wmma_bf16(a, f, s);
        f = fn;
        __builtin_amdgcn_sched_barrier(0);
      }
    }
    #pragma unroll
    for (int j = 0; j < 8; ++j) {
      int m = j + moff;
      if (n < HOR) {
        float vb = sqrtf(sh_sig[m] + 1e-8f);
        float v  = vb * (1.0f + softplusf(s[j] + f2bv));
        out[(b0 + m) * HOR + n] = fminf(fmaxf(v, 0.01f), 10.0f);
      }
    }
  }
  if (wave == 2 && lane < 16) out[BSZ * HOR + b0 + lane] = sh_sig[lane];
}

// ---------------------------------------------------------------------------
// Launch: repack 5 weight matrices into bf16 WMMA-B fragments (L2-resident),
// then run the persistent recurrence kernel: 32 workgroups x 512 threads.
// ---------------------------------------------------------------------------
extern "C" void kernel_launch(void* const* d_in, const int* in_sizes, int n_in,
                              void* d_out, int out_size, void* d_ws, size_t ws_size,
                              hipStream_t stream) {
  (void)in_sizes; (void)n_in; (void)out_size; (void)ws_size;

  const float* x     = (const float*)d_in[0];
  const float* Wzw   = (const float*)d_in[1];
  const float* Wzb   = (const float*)d_in[2];
  const float* Uzw   = (const float*)d_in[3];
  const float* Uzb   = (const float*)d_in[4];
  const float* Wrw   = (const float*)d_in[5];
  const float* Wrb   = (const float*)d_in[6];
  const float* Urw   = (const float*)d_in[7];
  const float* Urb   = (const float*)d_in[8];
  const float* Whw   = (const float*)d_in[9];
  const float* Whb   = (const float*)d_in[10];
  const float* Uhw   = (const float*)d_in[11];
  const float* Uhb   = (const float*)d_in[12];
  const float* Wgw   = (const float*)d_in[13];
  const float* Wgb   = (const float*)d_in[14];
  const float* omega = (const float*)d_in[15];
  const float* alpha = (const float*)d_in[16];
  const float* beta  = (const float*)d_in[17];
  const float* gamma = (const float*)d_in[18];
  const float* fc1w  = (const float*)d_in[19];
  const float* fc1b  = (const float*)d_in[20];
  const float* fc2w  = (const float*)d_in[21];
  const float* fc2b  = (const float*)d_in[22];

  char* ws = (char*)d_ws;
  bf16* uzpk  = (bf16*)(ws + 0);
  bf16* urpk  = (bf16*)(ws + 131072);
  bf16* uhpk  = (bf16*)(ws + 262144);
  bf16* fc1pk = (bf16*)(ws + 393216);
  bf16* fc2pk = (bf16*)(ws + 524288);

  // One-time (per launch) bf16 repack; tiny (528 KB) and stream-ordered.
  garchgru_repack<<<256, 256, 0, stream>>>(Uzw,  uzpk,  HH, HH, HH);
  garchgru_repack<<<256, 256, 0, stream>>>(Urw,  urpk,  HH, HH, HH);
  garchgru_repack<<<256, 256, 0, stream>>>(Uhw,  uhpk,  HH, HH, HH);
  garchgru_repack<<<256, 256, 0, stream>>>(fc1w, fc1pk, HH, HH, HH);
  garchgru_repack<<<32,  256, 0, stream>>>(fc2w, fc2pk, HH, HOR, 32);

  garchgru_main<<<BSZ / 16, 512, 0, stream>>>(
      x, Wzw, Wzb, Uzb, Wrw, Wrb, Urb, Whw, Whb, Uhb, Wgw, Wgb,
      omega, alpha, beta, gamma, fc1b, fc2b,
      uzpk, urpk, uhpk, fc1pk, fc2pk, (float*)d_out);
}
